// Model_8332236554274
// MI455X (gfx1250) — matrix operand order repaired and checked by reading, not compiled, not run on hardware
//
#include <hip/hip_runtime.h>
#include <hip/hip_bf16.h>

typedef __attribute__((ext_vector_type(16))) _Float16 v16h;
typedef __attribute__((ext_vector_type(8)))  _Float16 v8h;
typedef __attribute__((ext_vector_type(8)))  float    v8f;

#define E_ 512
#define H_ 512
#define C_ 10
#define B_ 256
#define T_ 64

__device__ __forceinline__ float sigmoidf_(float x) {
    return 1.0f / (1.0f + __expf(-x));
}
__device__ __forceinline__ float seluf_(float x) {
    const float sc = 1.0507009873554805f;      // selu scale
    const float sa = 1.7580993408473766f;      // scale * alpha
    return x > 0.0f ? sc * x : sa * (__expf(x) - 1.0f);
}

// ---------------------------------------------------------------------------
// Prep: fold [Wx ; Wh] (each E/H x 4H, row-major f32) into transposed f16
// WT[n][k], n in [0,2048), k in [0,1024): k<512 -> Wx[k][n], else Wh[k-512][n].
// Makes WMMA B-fragment loads contiguous 32B chunks per lane.
// ---------------------------------------------------------------------------
__global__ void prep_w(const float* __restrict__ Wx, const float* __restrict__ Wh,
                       _Float16* __restrict__ WT) {
    size_t i = (size_t)blockIdx.x * 256 + threadIdx.x;       // over 2048*1024
    if (i >= (size_t)2048 * 1024) return;
    int k = (int)(i & 1023);
    int n = (int)(i >> 10);
    float v = (k < 512) ? Wx[(size_t)k * 2048 + n]
                        : Wh[(size_t)(k - 512) * 2048 + n];
    WT[i] = (_Float16)v;
}

// ---------------------------------------------------------------------------
// Embedding gather -> time-major f16 activations: xbf[t][b][e]
// ---------------------------------------------------------------------------
__global__ void gather_x(const int* __restrict__ tokens, const float* __restrict__ emb,
                         _Float16* __restrict__ xbf) {
    size_t i = (size_t)blockIdx.x * 256 + threadIdx.x;       // over T*B*E
    if (i >= (size_t)T_ * B_ * E_) return;
    int e = (int)(i & 511);
    int b = (int)((i >> 9) & 255);
    int t = (int)(i >> 17);
    int tok = tokens[b * T_ + t];
    xbf[i] = (_Float16)emb[(size_t)tok * E_ + e];
}

// ---------------------------------------------------------------------------
// Main persistent LIAF-LSTM recurrence.
// Grid: 32 blocks = 2 directions x 16 batch-tiles (16 rows each), 256 thr/blk.
// Each wave owns 64 hidden units across all 4 gate slices -> gate combine is
// register-local. h double-buffered in LDS (one barrier/step).
// An opaque zero OFFSET (not a laundered pointer!) is added per step: this
// blocks LICM of the loop-invariant weight loads (which previously spilled
// 16KB/lane to scratch) while preserving the kernarg pointer's global
// address-space provenance, so weight loads stay global_load_b128 (LOADcnt
// only) and never couple with the LDS DScnt pipe the way flat_load does.
// ---------------------------------------------------------------------------
__global__ __launch_bounds__(256, 1)
void liaf_main(const _Float16* __restrict__ xbf,
               const _Float16* __restrict__ WT_f, const _Float16* __restrict__ WT_b,
               const float* __restrict__ b_f, const float* __restrict__ b_b,
               _Float16* __restrict__ hseq_f, _Float16* __restrict__ hseq_b) {
    __shared__ __align__(32) _Float16 h_lds[2][16][H_];      // 32 KB, double-buffered

    const int tid  = threadIdx.x;
    const int lane = tid & 31;
    const int wave = tid >> 5;
    const int dir  = blockIdx.x >> 4;
    const int brow = (blockIdx.x & 15) * 16;                  // batch tile base

    const _Float16* __restrict__ WT   = dir ? WT_b  : WT_f;
    const float*    __restrict__ bias = dir ? b_b   : b_f;
    _Float16*       __restrict__ hseq = dir ? hseq_b : hseq_f;

    const int lm = lane & 15;     // A/C row within tile; B/C column within tile
    const int lh = lane >> 4;     // half-wave select

    // zero initial hidden state (buffer 0)
    for (int idx = tid; idx < 16 * H_; idx += 256)
        (&h_lds[0][0][0])[idx] = (_Float16)0.0f;
    __syncthreads();

    // per-lane bias for each (gate g, sub-tile s); column-indexed
    float bia[4][4];
#pragma unroll
    for (int g = 0; g < 4; ++g)
#pragma unroll
        for (int s = 0; s < 4; ++s)
            bia[g][s] = bias[g * 512 + wave * 64 + s * 16 + lm];

    // per-lane element offset of this lane's column base + K-chunk select.
    // B frag addr = WT + cbase + (g*512 + s*16)*1024 (imm) + koff + kt*32
    const size_t cbase = (size_t)(wave * 64 + lm) * 1024 + (size_t)(lh * 16);

    float creg[4][8];
#pragma unroll
    for (int s = 0; s < 4; ++s)
#pragma unroll
        for (int r = 0; r < 8; ++r) creg[s][r] = 0.0f;

    for (int t = 0; t < T_; ++t) {
        // opaque ZERO offset, redefined every iteration: blocks LICM of the
        // weight loads across the t-loop without destroying the pointer's
        // global address-space provenance.
        unsigned long long zo = 0;
        asm volatile("" : "+s"(zo));
        const _Float16* WTl = WT + (cbase + zo);

        const int cur = t & 1;
        const int nxt = cur ^ 1;
        const int xt  = dir ? (T_ - 1 - t) : t;
        const _Float16* xrow = xbf + ((size_t)xt * B_ + brow + lm) * E_;

#pragma unroll
        for (int half = 0; half < 2; ++half) {
            v8f acc[8];
            v8f zv = {};
#pragma unroll
            for (int i = 0; i < 8; ++i) acc[i] = zv;

            // --- K = 0..511 : x_t from global (L2-resident) ---
            for (int kt = 0; kt < 16; ++kt) {
                union { v16h v; v8h h[2]; } a;
                const _Float16* p = xrow + kt * 32 + lh * 16;
                a.h[0] = *(const v8h*)p;
                a.h[1] = *(const v8h*)(p + 8);
#pragma unroll
                for (int g = 0; g < 4; ++g)
#pragma unroll
                    for (int s2 = 0; s2 < 2; ++s2) {
                        const int s = half * 2 + s2;
                        v16h bf = *(const v16h*)(WTl + (size_t)((g * 512 + s * 16) * 1024)
                                                     + kt * 32);
                        acc[g * 2 + s2] = __builtin_amdgcn_wmma_f32_16x16x32_f16(
                            false, a.v, false, bf, (short)0, acc[g * 2 + s2],
                            false, false);
                    }
            }
            // --- K = 512..1023 : h_{t-1} from LDS (current buffer) ---
            for (int kt = 0; kt < 16; ++kt) {
                union { v16h v; v8h h[2]; } a;
                const _Float16* p = &h_lds[cur][lm][kt * 32 + lh * 16];
                a.h[0] = *(const v8h*)p;
                a.h[1] = *(const v8h*)(p + 8);
#pragma unroll
                for (int g = 0; g < 4; ++g)
#pragma unroll
                    for (int s2 = 0; s2 < 2; ++s2) {
                        const int s = half * 2 + s2;
                        v16h bf = *(const v16h*)(WTl + (size_t)((g * 512 + s * 16) * 1024)
                                                     + 512 + kt * 32);
                        acc[g * 2 + s2] = __builtin_amdgcn_wmma_f32_16x16x32_f16(
                            false, a.v, false, bf, (short)0, acc[g * 2 + s2],
                            false, false);
                    }
            }

            // gate combine: all 4 gate slices of unit j live in this wave.
            // Write h_new straight into the next LDS buffer (no extra regs).
#pragma unroll
            for (int s2 = 0; s2 < 2; ++s2) {
                const int s = half * 2 + s2;
#pragma unroll
                for (int r = 0; r < 8; ++r) {
                    float zi = acc[0 * 2 + s2][r] + bia[0][s];
                    float zf = acc[1 * 2 + s2][r] + bia[1][s];
                    float zg = acc[2 * 2 + s2][r] + bia[2][s];
                    float zo2 = acc[3 * 2 + s2][r] + bia[3][s];
                    float ii = sigmoidf_(zi);
                    float ff = sigmoidf_(zf);
                    float oo = sigmoidf_(zo2);
                    float gg = seluf_(zg);
                    float cn = 0.5f * ff * creg[s][r] + ii * gg;   // DECAY=0.5
                    float hn = oo * seluf_(cn);
                    creg[s][r] = (cn > 0.5f) ? 0.0f : cn;          // fire/reset
                    const int M = r + lh * 8;                      // C-layout row
                    const int j = wave * 64 + s * 16 + lm;         // hidden unit
                    _Float16 hv = (_Float16)hn;
                    h_lds[nxt][M][j] = hv;
                    hseq[((size_t)t * B_ + brow + M) * H_ + j] = hv;
                }
            }
        }

        if (t + 1 < T_) {
            const int xn = dir ? (T_ - 2 - t) : (t + 1);
            __builtin_prefetch(xbf + ((size_t)xn * B_ + brow + lm) * E_, 0, 1);
        }
        // one barrier: step-t writes to h_lds[nxt] visible, and step-t reads of
        // h_lds[cur] finished, before buffers swap roles next iteration.
        __syncthreads();
    }
}

// ---------------------------------------------------------------------------
// Output head: per-batch-row membrane recurrence over T with W_fc (1024 x 10).
// One wave per batch row; W_fc staged in LDS; butterfly reduce across lanes.
// ---------------------------------------------------------------------------
__global__ __launch_bounds__(32)
void fc_out(const _Float16* __restrict__ hf_seq, const _Float16* __restrict__ hb_seq,
            const float* __restrict__ Wfc, const float* __restrict__ bfc,
            float* __restrict__ out) {
    __shared__ float Wl[1024 * C_];                 // 40 KB
    const int b = blockIdx.x;
    const int lane = threadIdx.x;
    for (int i = lane; i < 1024 * C_; i += 32) Wl[i] = Wfc[i];
    __syncthreads();

    float v[C_], accv[C_];
#pragma unroll
    for (int c = 0; c < C_; ++c) { v[c] = 0.0f; accv[c] = 0.0f; }

    for (int t = 0; t < T_; ++t) {
        const _Float16* hf = hf_seq + ((size_t)t * B_ + b) * H_;
        const _Float16* hb = hb_seq + ((size_t)t * B_ + b) * H_;
        float xf[16], xb[16];
#pragma unroll
        for (int q = 0; q < 16; ++q) {
            xf[q] = (float)hf[lane + 32 * q];
            xb[q] = (float)hb[lane + 32 * q];
        }
#pragma unroll
        for (int c = 0; c < C_; ++c) {
            float p = 0.0f;
#pragma unroll
            for (int q = 0; q < 16; ++q) {
                int j = lane + 32 * q;
                p += xf[q] * Wl[j * C_ + c] + xb[q] * Wl[(512 + j) * C_ + c];
            }
            for (int m = 16; m > 0; m >>= 1) p += __shfl_xor(p, m, 32);
            float u = p + bfc[c] + 0.5f * v[c];                    // DECAY*v
            v[c] = (u > 0.5f) ? 0.0f : u;                          // fire/reset
            accv[c] += seluf_(u);
        }
    }
    if (lane == 0)
#pragma unroll
        for (int c = 0; c < C_; ++c)
            out[b * C_ + c] = accv[c] * (1.0f / (float)T_);
}

// ---------------------------------------------------------------------------
extern "C" void kernel_launch(void* const* d_in, const int* in_sizes, int n_in,
                              void* d_out, int out_size, void* d_ws, size_t ws_size,
                              hipStream_t stream) {
    const int*   tokens = (const int*)d_in[0];
    const float* emb    = (const float*)d_in[1];
    const float* Wx_f   = (const float*)d_in[2];
    const float* Wh_f   = (const float*)d_in[3];
    const float* b_f    = (const float*)d_in[4];
    const float* Wx_b   = (const float*)d_in[5];
    const float* Wh_b   = (const float*)d_in[6];
    const float* b_b    = (const float*)d_in[7];
    const float* W_fc   = (const float*)d_in[8];
    const float* b_fc   = (const float*)d_in[9];
    float* out = (float*)d_out;

    // Workspace layout (bytes):
    //   xbf   : T*B*E f16          = 16,777,216 (33,554,432 reserved)
    //   WT_f  : 2048*1024 f16      =  4,194,304
    //   WT_b  : 2048*1024 f16      =  4,194,304
    //   hseq_f: T*B*H f16          = 16,777,216
    //   hseq_b: T*B*H f16          = 16,777,216   (total ~75.5 MB)
    char* ws = (char*)d_ws;
    _Float16* xbf  = (_Float16*)(ws);
    _Float16* WT_f = (_Float16*)(ws + 33554432ull);
    _Float16* WT_b = (_Float16*)(ws + 37748736ull);
    _Float16* hsf  = (_Float16*)(ws + 41943040ull);
    _Float16* hsb  = (_Float16*)(ws + 58720256ull);

    const int nW = (2048 * 1024 + 255) / 256;
    prep_w<<<nW, 256, 0, stream>>>(Wx_f, Wh_f, WT_f);
    prep_w<<<nW, 256, 0, stream>>>(Wx_b, Wh_b, WT_b);

    const int nX = (int)(((size_t)T_ * B_ * E_ + 255) / 256);
    gather_x<<<nX, 256, 0, stream>>>(tokens, emb, xbf);

    liaf_main<<<32, 256, 0, stream>>>(xbf, WT_f, WT_b, b_f, b_b, hsf, hsb);

    fc_out<<<B_, 32, 0, stream>>>(hsf, hsb, W_fc, b_fc, out);
}
